// RENet_81260781240937
// MI455X (gfx1250) — compile-verified, matrix operand order checked by reading, not run on hardware
//
#include <hip/hip_runtime.h>
#include <hip/hip_bf16.h>

typedef __bf16 bf16_t;
typedef __attribute__((ext_vector_type(16))) __bf16 v16bf;
typedef __attribute__((ext_vector_type(8)))  __bf16 v8bf;
typedef __attribute__((ext_vector_type(8)))  float   v8f;

#define HDIM   200
#define SEQLEN 10
#define BATCH  1024
#define NEDGE  65536
#define NNODES 50000
#define K3     600
#define KP     608   // 600 padded to 19*32
#define HP     224   // 200 padded to 7*32
#define KCH    256   // K chunk staged in LDS (8 WMMA steps)

// ---------------- utility kernels ----------------

__global__ void fill_zero(float* __restrict__ p, size_t n) {
    size_t i = (size_t)blockIdx.x * blockDim.x + threadIdx.x;
    if (i < n) p[i] = 0.0f;
}

// scatter-add of ent rows into per-(batch,t) slots + counts
__global__ void scatter_add(const float* __restrict__ ent, const int* __restrict__ nid,
                            const int* __restrict__ tt, const int* __restrict__ bb,
                            float* __restrict__ sum, float* __restrict__ cnt) {
    const int e    = blockIdx.x;
    const int node = nid[e];
    const int slot = tt[e] + bb[e] * SEQLEN;
    const int h    = threadIdx.x;
    if (h < HDIM)
        atomicAdd(&sum[(size_t)slot * HDIM + h], ent[(size_t)node * HDIM + h]);
    if (h == 0)
        atomicAdd(&cnt[slot], 1.0f);
}

__global__ void normalize_mean(float* __restrict__ sum, const float* __restrict__ cnt) {
    const int i = blockIdx.x;
    const int h = threadIdx.x;
    if (h < HDIM) sum[(size_t)i * HDIM + h] /= fmaxf(cnt[i], 1.0f);
}

// f32 [N,K] -> bf16 [Np,Kp] zero-padded
__global__ void convert_pad(const float* __restrict__ W, bf16_t* __restrict__ Wp,
                            int N, int K, int Np, int Kp) {
    int i = blockIdx.x * blockDim.x + threadIdx.x;
    if (i >= Np * Kp) return;
    int n = i / Kp, k = i % Kp;
    float v = (n < N && k < K) ? W[(size_t)n * K + k] : 0.0f;
    Wp[i] = (bf16_t)v;
}

// x_t = concat(e_idx, mean_t, e_rel) -> bf16 [B, KP]
__global__ void build_x(const float* __restrict__ ent, const float* __restrict__ remb,
                        const int* __restrict__ idx, const int* __restrict__ rel,
                        const float* __restrict__ mean, int t, bf16_t* __restrict__ x) {
    const int b = blockIdx.x;
    for (int k = threadIdx.x; k < KP; k += blockDim.x) {
        float v = 0.0f;
        if (k < HDIM)            v = ent [(size_t)idx[b] * HDIM + k];
        else if (k < 2 * HDIM)   v = mean[((size_t)b * SEQLEN + t) * HDIM + (k - HDIM)];
        else if (k < 3 * HDIM)   v = remb[(size_t)rel[b] * HDIM + (k - 2 * HDIM)];
        x[(size_t)b * KP + k] = (bf16_t)v;
    }
}

// f = concat(e_idx, hT, e_rel) -> bf16 [B, KP]
__global__ void build_f(const float* __restrict__ ent, const float* __restrict__ remb,
                        const int* __restrict__ idx, const int* __restrict__ rel,
                        const float* __restrict__ h, bf16_t* __restrict__ f) {
    const int b = blockIdx.x;
    for (int k = threadIdx.x; k < KP; k += blockDim.x) {
        float v = 0.0f;
        if (k < HDIM)            v = ent [(size_t)idx[b] * HDIM + k];
        else if (k < 2 * HDIM)   v = h   [(size_t)b * HDIM + (k - HDIM)];
        else if (k < 3 * HDIM)   v = remb[(size_t)rel[b] * HDIM + (k - 2 * HDIM)];
        f[(size_t)b * KP + k] = (bf16_t)v;
    }
}

// fp32 hidden [B,H] -> bf16 [B,HP] zero-padded
__global__ void convert_h(const float* __restrict__ h, bf16_t* __restrict__ hb) {
    const int b = blockIdx.x;
    for (int k = threadIdx.x; k < HP; k += blockDim.x) {
        float v = (k < HDIM) ? h[(size_t)b * HDIM + k] : 0.0f;
        hb[(size_t)b * HP + k] = (bf16_t)v;
    }
}

// GRU gate math: h <- (1-z)*n + z*h  (biases added here)
__global__ void gru_gate(float* __restrict__ h, const float* __restrict__ gi,
                         const float* __restrict__ gh,
                         const float* __restrict__ b_ih, const float* __restrict__ b_hh) {
    const int b = blockIdx.x;
    const int j = threadIdx.x;
    if (j >= HDIM) return;
    const float* gib = gi + (size_t)b * KP;
    const float* ghb = gh + (size_t)b * KP;
    float ir = gib[j]            + b_ih[j];
    float iz = gib[HDIM + j]     + b_ih[HDIM + j];
    float in_= gib[2 * HDIM + j] + b_ih[2 * HDIM + j];
    float hr = ghb[j]            + b_hh[j];
    float hz = ghb[HDIM + j]     + b_hh[HDIM + j];
    float hn = ghb[2 * HDIM + j] + b_hh[2 * HDIM + j];
    float r = 1.0f / (1.0f + expf(-(ir + hr)));
    float z = 1.0f / (1.0f + expf(-(iz + hz)));
    float n = tanhf(in_ + r * hn);
    float* hp = h + (size_t)b * HDIM + j;
    *hp = (1.0f - z) * n + z * (*hp);
}

// ---------------- WMMA GEMMs ----------------
// A fragment (16x32 bf16): lane<16 -> row M=lane, K {k0..k0+7, k0+16..k0+23};
// lane>=16 -> row M=lane-16, K {k0+8..k0+15, k0+24..k0+31}. Two 16B loads.
__device__ __forceinline__ v16bf load_frag(const bf16_t* row, int k0, int half) {
    v8bf lo = *(const v8bf*)(row + k0 + half * 8);
    v8bf hi = *(const v8bf*)(row + k0 + 16 + half * 8);
    return __builtin_shufflevector(lo, hi, 0,1,2,3,4,5,6,7,8,9,10,11,12,13,14,15);
}

// C[M,N] = A[M,Kp] * W[N,Kp]^T ; one wave = one 16x16 tile; stores with ldc
__global__ void __launch_bounds__(32)
gemm_bf16_16x16(const bf16_t* __restrict__ A, const bf16_t* __restrict__ W,
                float* __restrict__ C, int Kp, int ldc) {
    const int m0 = blockIdx.x * 16;
    const int n0 = blockIdx.y * 16;
    const int lane = threadIdx.x;
    const int half = lane >> 4, r = lane & 15;
    const bf16_t* arow = A + (size_t)(m0 + r) * Kp;
    const bf16_t* wrow = W + (size_t)(n0 + r) * Kp;
    v8f acc = {};
    for (int k0 = 0; k0 < Kp; k0 += 32) {
        v16bf a = load_frag(arow, k0, half);
        v16bf b = load_frag(wrow, k0, half);
        acc = __builtin_amdgcn_wmma_f32_16x16x32_bf16(false, a, false, b,
                                                      (short)0, acc, false, false);
    }
    float* cp = C + (size_t)(m0 + half * 8) * ldc + (n0 + r);
#pragma unroll
    for (int i = 0; i < 8; ++i) cp[(size_t)i * ldc] = acc[i];
}

// logits: A bf16 [B,KP] x W_lin f32 [NNODES,K3] -> C f32 [B,NNODES] (+bias)
// 8-wave workgroup computes a 128x80 block: the 80xK weight strip is converted
// to bf16 ONCE and staged in LDS (40KB chunks); all 8 waves consume it via
// ds_load_b128, amortizing fp32 W fetch + conversion 8x.
__global__ void __launch_bounds__(256)
gemm_logits(const bf16_t* __restrict__ A, const float* __restrict__ W,
            const float* __restrict__ bias, float* __restrict__ C) {
    __shared__ bf16_t smem[80 * KCH];                 // 40,960 B

    const int wave = threadIdx.x >> 5;                // 0..7
    const int lane = threadIdx.x & 31;
    const int half = lane >> 4, r = lane & 15;
    const int m0 = blockIdx.x * 128 + wave * 16;      // per-wave M tile
    const int n0 = blockIdx.y * 80;                   // shared N strip

    const bf16_t* arow = A + (size_t)(m0 + r) * KP;
    v8f acc[5] = {{}, {}, {}, {}, {}};

    for (int kb = 0; kb < KP; kb += KCH) {
        const int klen = (KP - kb) < KCH ? (KP - kb) : KCH;   // 256,256,96 (mult of 32)
        const int ng = klen >> 3;                              // 8-float groups per row
        // cooperative stage: fp32 -> bf16 into LDS
        for (int g = threadIdx.x; g < 80 * ng; g += 256) {
            const int rrow = g / ng;
            const int kg   = (g - rrow * ng) << 3;
            const int kglo = kb + kg;
            v8f v = {};
            if (kglo < K3) {
                const float* wp = W + (size_t)(n0 + rrow) * K3 + kglo;
                v = *(const v8f*)wp;
                __builtin_prefetch(wp + KCH, 0, 1);            // next K chunk
            }
            v8bf bb;
#pragma unroll
            for (int i = 0; i < 8; ++i) bb[i] = (bf16_t)v[i];
            *(v8bf*)(&smem[rrow * KCH + kg]) = bb;
        }
        __syncthreads();

        for (int k0 = 0; k0 < klen; k0 += 32) {
            v16bf a = load_frag(arow, kb + k0, half);
#pragma unroll
            for (int t = 0; t < 5; ++t) {
                const int wbase = (t * 16 + r) * KCH;
                v8bf lo = *(const v8bf*)(&smem[wbase + k0 + half * 8]);
                v8bf hi = *(const v8bf*)(&smem[wbase + k0 + 16 + half * 8]);
                v16bf b = __builtin_shufflevector(lo, hi,
                          0,1,2,3,4,5,6,7,8,9,10,11,12,13,14,15);
                acc[t] = __builtin_amdgcn_wmma_f32_16x16x32_bf16(false, a, false, b,
                                                                 (short)0, acc[t],
                                                                 false, false);
            }
        }
        __syncthreads();
    }

#pragma unroll
    for (int t = 0; t < 5; ++t) {
        const int n = n0 + t * 16 + r;
        const float bv = bias[n];
        float* cp = C + (size_t)(m0 + half * 8) * (size_t)NNODES + n;
#pragma unroll
        for (int i = 0; i < 8; ++i) cp[(size_t)i * NNODES] = acc[t][i] + bv;
    }
}

// in-place row log-softmax, one block per row
__global__ void logsoftmax_rows(float* __restrict__ data, int cols) {
    const int row = blockIdx.x;
    float* p = data + (size_t)row * cols;
    __shared__ float red[256];
    const int tid = threadIdx.x;

    float m = -3.402823466e+38f;
    for (int c = tid; c < cols; c += blockDim.x) m = fmaxf(m, p[c]);
    red[tid] = m; __syncthreads();
    for (int s = 128; s > 0; s >>= 1) {
        if (tid < s) red[tid] = fmaxf(red[tid], red[tid + s]);
        __syncthreads();
    }
    m = red[0]; __syncthreads();

    float sum = 0.0f;
    for (int c = tid; c < cols; c += blockDim.x) sum += expf(p[c] - m);
    red[tid] = sum; __syncthreads();
    for (int s = 128; s > 0; s >>= 1) {
        if (tid < s) red[tid] += red[tid + s];
        __syncthreads();
    }
    const float lse = m + logf(red[0]);
    __syncthreads();

    for (int c = tid; c < cols; c += blockDim.x) p[c] -= lse;
}

// ---------------- orchestration ----------------

extern "C" void kernel_launch(void* const* d_in, const int* in_sizes, int n_in,
                              void* d_out, int out_size, void* d_ws, size_t ws_size,
                              hipStream_t stream) {
    (void)in_sizes; (void)n_in; (void)out_size; (void)ws_size;
    const int*   sub       = (const int*)  d_in[0];
    const int*   rel       = (const int*)  d_in[1];
    const int*   obj       = (const int*)  d_in[2];
    const int*   h_sub     = (const int*)  d_in[3];
    const int*   h_sub_t   = (const int*)  d_in[4];
    const int*   h_sub_b   = (const int*)  d_in[5];
    const int*   h_obj     = (const int*)  d_in[6];
    const int*   h_obj_t   = (const int*)  d_in[7];
    const int*   h_obj_b   = (const int*)  d_in[8];
    const float* ent       = (const float*)d_in[9];
    const float* rel_emb   = (const float*)d_in[10];
    const float* W_ih_sub  = (const float*)d_in[11];
    const float* W_hh_sub  = (const float*)d_in[12];
    const float* b_ih_sub  = (const float*)d_in[13];
    const float* b_hh_sub  = (const float*)d_in[14];
    const float* W_ih_obj  = (const float*)d_in[15];
    const float* W_hh_obj  = (const float*)d_in[16];
    const float* b_ih_obj  = (const float*)d_in[17];
    const float* b_hh_obj  = (const float*)d_in[18];
    const float* W_lin_sub = (const float*)d_in[19];
    const float* b_lin_sub = (const float*)d_in[20];
    const float* W_lin_obj = (const float*)d_in[21];
    const float* b_lin_obj = (const float*)d_in[22];

    char* w = (char*)d_ws;
    size_t off = 0;
    auto carve = [&](size_t bytes) -> void* {
        void* p = w + off;
        off = (off + bytes + 255) & ~(size_t)255;
        return p;
    };
    float*  hs_sum   = (float*) carve((size_t)BATCH * SEQLEN * HDIM * 4);
    float*  hs_cnt   = (float*) carve((size_t)BATCH * SEQLEN * 4);
    float*  ho_sum   = (float*) carve((size_t)BATCH * SEQLEN * HDIM * 4);
    float*  ho_cnt   = (float*) carve((size_t)BATCH * SEQLEN * 4);
    float*  hst_sub  = (float*) carve((size_t)BATCH * HDIM * 4);
    float*  hst_obj  = (float*) carve((size_t)BATCH * HDIM * 4);
    const size_t zero_span = off;           // everything above zeroed each call
    float*  gi       = (float*) carve((size_t)BATCH * KP * 4);
    float*  gh       = (float*) carve((size_t)BATCH * KP * 4);
    bf16_t* xb       = (bf16_t*)carve((size_t)BATCH * KP * 2);
    bf16_t* hb       = (bf16_t*)carve((size_t)BATCH * HP * 2);
    bf16_t* f_sub    = (bf16_t*)carve((size_t)BATCH * KP * 2);
    bf16_t* f_obj    = (bf16_t*)carve((size_t)BATCH * KP * 2);
    bf16_t* Wih_sub_p= (bf16_t*)carve((size_t)KP * KP * 2);
    bf16_t* Wih_obj_p= (bf16_t*)carve((size_t)KP * KP * 2);
    bf16_t* Whh_sub_p= (bf16_t*)carve((size_t)KP * HP * 2);
    bf16_t* Whh_obj_p= (bf16_t*)carve((size_t)KP * HP * 2);

    // zero accumulators + GRU states (determinism across replays)
    {
        size_t nz = zero_span / 4;
        fill_zero<<<(unsigned)((nz + 255) / 256), 256, 0, stream>>>((float*)d_ws, nz);
    }

    // pad/convert GRU weights to bf16
    {
        int tot = KP * KP;
        convert_pad<<<(tot + 255) / 256, 256, 0, stream>>>(W_ih_sub, Wih_sub_p, K3, K3, KP, KP);
        convert_pad<<<(tot + 255) / 256, 256, 0, stream>>>(W_ih_obj, Wih_obj_p, K3, K3, KP, KP);
        tot = KP * HP;
        convert_pad<<<(tot + 255) / 256, 256, 0, stream>>>(W_hh_sub, Whh_sub_p, K3, HDIM, KP, HP);
        convert_pad<<<(tot + 255) / 256, 256, 0, stream>>>(W_hh_obj, Whh_obj_p, K3, HDIM, KP, HP);
    }

    // scatter-mean histories
    scatter_add<<<NEDGE, 224, 0, stream>>>(ent, h_sub, h_sub_t, h_sub_b, hs_sum, hs_cnt);
    scatter_add<<<NEDGE, 224, 0, stream>>>(ent, h_obj, h_obj_t, h_obj_b, ho_sum, ho_cnt);
    normalize_mean<<<BATCH * SEQLEN, 224, 0, stream>>>(hs_sum, hs_cnt);
    normalize_mean<<<BATCH * SEQLEN, 224, 0, stream>>>(ho_sum, ho_cnt);

    // GRUs (WMMA for both GEMMs each step)
    dim3 g_gemm(BATCH / 16, KP / 16);   // 64 x 38 tiles, one wave each
    for (int t = 0; t < SEQLEN; ++t) {
        build_x<<<BATCH, 256, 0, stream>>>(ent, rel_emb, sub, rel, hs_sum, t, xb);
        gemm_bf16_16x16<<<g_gemm, 32, 0, stream>>>(xb, Wih_sub_p, gi, KP, KP);
        convert_h<<<BATCH, 256, 0, stream>>>(hst_sub, hb);
        gemm_bf16_16x16<<<g_gemm, 32, 0, stream>>>(hb, Whh_sub_p, gh, HP, KP);
        gru_gate<<<BATCH, 224, 0, stream>>>(hst_sub, gi, gh, b_ih_sub, b_hh_sub);
    }
    for (int t = 0; t < SEQLEN; ++t) {
        build_x<<<BATCH, 256, 0, stream>>>(ent, rel_emb, obj, rel, ho_sum, t, xb);
        gemm_bf16_16x16<<<g_gemm, 32, 0, stream>>>(xb, Wih_obj_p, gi, KP, KP);
        convert_h<<<BATCH, 256, 0, stream>>>(hst_obj, hb);
        gemm_bf16_16x16<<<g_gemm, 32, 0, stream>>>(hb, Whh_obj_p, gh, HP, KP);
        gru_gate<<<BATCH, 224, 0, stream>>>(hst_obj, gi, gh, b_ih_obj, b_hh_obj);
    }

    // features + classifier GEMMs straight into d_out
    build_f<<<BATCH, 256, 0, stream>>>(ent, rel_emb, sub, rel, hst_sub, f_sub);
    build_f<<<BATCH, 256, 0, stream>>>(ent, rel_emb, obj, rel, hst_obj, f_obj);

    float* out_obj = (float*)d_out;                              // log_prob_obj
    float* out_sub = out_obj + (size_t)BATCH * NNODES;           // log_prob_sub
    dim3 g_big(BATCH / 128, NNODES / 80);                        // 8 x 625 blocks
    gemm_logits<<<g_big, 256, 0, stream>>>(f_sub, W_lin_sub, b_lin_sub, out_obj);
    gemm_logits<<<g_big, 256, 0, stream>>>(f_obj, W_lin_obj, b_lin_obj, out_sub);

    // in-place log-softmax over all 2048 rows
    logsoftmax_rows<<<2 * BATCH, 256, 0, stream>>>((float*)d_out, NNODES);
}
